// GraphConv_43207370998364
// MI455X (gfx1250) — compile-verified, matrix operand order checked by reading, not run on hardware
//
#include <hip/hip_runtime.h>

#define N_NODES 131072
#define N_EDGES 4194304
#define DIM     128

typedef __attribute__((ext_vector_type(16))) __bf16 v16bf;
typedef __attribute__((ext_vector_type(8)))  float  v8f;

// ---------------------------------------------------------------------------
// Kernel 0: zero the pre_embed scratch (N*D floats) with float4 stores.
// ---------------------------------------------------------------------------
__global__ __launch_bounds__(256) void gc_zero(float4* __restrict__ p) {
    size_t i = (size_t)blockIdx.x * 256 + threadIdx.x;
    p[i] = make_float4(0.f, 0.f, 0.f, 0.f);
}

// ---------------------------------------------------------------------------
// Kernel 1: SpMM scatter.  One wave32 per edge; lane l handles columns
// [4l, 4l+4).  x row gather is one contiguous 512B read per wave (float4 per
// lane), accumulation via global_atomic_add_f32 (L2-resident, 192MB L2 holds
// both x and pre_embed).
// ---------------------------------------------------------------------------
__global__ __launch_bounds__(256) void gc_spmm_scatter(
    const float* __restrict__ x,
    const int*   __restrict__ erow,
    const int*   __restrict__ ecol,
    const float* __restrict__ eval,
    float*       __restrict__ pre)
{
    const int lane = threadIdx.x & 31;
    const int wave = threadIdx.x >> 5;
    const long long e = (long long)blockIdx.x * 8 + wave;   // grid = E/8

    const int   r = erow[e];          // wave-uniform -> scalar loads
    const int   c = ecol[e];
    const float v = eval[e];

    const float4 xv = *(const float4*)(x + (long long)c * DIM + lane * 4);
    float* dst = pre + (long long)r * DIM + lane * 4;
    atomicAdd(dst + 0, xv.x * v);
    atomicAdd(dst + 1, xv.y * v);
    atomicAdd(dst + 2, xv.z * v);
    atomicAdd(dst + 3, xv.w * v);
}

// ---------------------------------------------------------------------------
// Kernel 2: out = pre_embed @ W^T via v_wmma_f32_16x16x32_bf16.
//
// Block = 256 threads = 8 waves, covers 128 output rows (16 per wave).
// W (128x128 f32) is staged once per block into LDS as bf16 B-fragments in
// native WMMA layout:  fragment (nt, kc) holds B = W^T[kc*32 .. +32, nt*16
// .. +16], i.e. b[k][n] = W[n][k].  Per ISA 7.12.2, lane L of a 16-bit B
// operand holds column n = L%16 and K elements {c0..c0+7, c0+16..c0+23}
// with c0 = (L>=16 ? 8 : 0) within the 32-K chunk.
// ---------------------------------------------------------------------------
__global__ __launch_bounds__(256) void gc_gemm_wmma(
    const float* __restrict__ pre,
    const float* __restrict__ W,
    float*       __restrict__ out)
{
    __shared__ v16bf bfrag[8 * 4 * 32];   // [ntile][kchunk][lane] = 32 KB

    const int lane = threadIdx.x & 31;
    const int wave = threadIdx.x >> 5;

    // ---- Stage W -> LDS as pre-swizzled bf16 B fragments (4 sets per wave)
    #pragma unroll
    for (int i = 0; i < 4; ++i) {
        const int set = wave * 4 + i;             // 0..31 = (nt,kc)
        const int nt  = set >> 2;
        const int kc  = set & 3;
        const int n   = nt * 16 + (lane & 15);
        const int c0  = kc * 32 + ((lane >> 4) << 3);
        const float* wrow = W + n * DIM;
        v16bf b;
        #pragma unroll
        for (int j = 0; j < 8; ++j) b[j]     = (__bf16)wrow[c0 + j];
        #pragma unroll
        for (int j = 0; j < 8; ++j) b[8 + j] = (__bf16)wrow[c0 + 16 + j];
        bfrag[set * 32 + lane] = b;
    }
    __syncthreads();

    // ---- Load this wave's 4 A fragments (16 rows x full K=128), bf16.
    // A layout: lane L holds row m0 + L%16; K offset c0 as for B.
    const long long m0 = ((long long)blockIdx.x * 8 + wave) * 16;
    const float* arow = pre + (m0 + (lane & 15)) * DIM;
    v16bf a[4];
    #pragma unroll
    for (int kc = 0; kc < 4; ++kc) {
        const int c0 = kc * 32 + ((lane >> 4) << 3);
        const float4 p0 = *(const float4*)(arow + c0);
        const float4 p1 = *(const float4*)(arow + c0 + 4);
        const float4 p2 = *(const float4*)(arow + c0 + 16);
        const float4 p3 = *(const float4*)(arow + c0 + 20);
        a[kc][0]  = (__bf16)p0.x; a[kc][1]  = (__bf16)p0.y;
        a[kc][2]  = (__bf16)p0.z; a[kc][3]  = (__bf16)p0.w;
        a[kc][4]  = (__bf16)p1.x; a[kc][5]  = (__bf16)p1.y;
        a[kc][6]  = (__bf16)p1.z; a[kc][7]  = (__bf16)p1.w;
        a[kc][8]  = (__bf16)p2.x; a[kc][9]  = (__bf16)p2.y;
        a[kc][10] = (__bf16)p2.z; a[kc][11] = (__bf16)p2.w;
        a[kc][12] = (__bf16)p3.x; a[kc][13] = (__bf16)p3.y;
        a[kc][14] = (__bf16)p3.z; a[kc][15] = (__bf16)p3.w;
    }

    // ---- 8 N-tiles x 4 K-chunks = 32 WMMAs per wave
    #pragma unroll
    for (int nt = 0; nt < 8; ++nt) {
        v8f acc = {};
        #pragma unroll
        for (int kc = 0; kc < 4; ++kc) {
            const v16bf b = bfrag[(nt * 4 + kc) * 32 + lane];
            acc = __builtin_amdgcn_wmma_f32_16x16x32_bf16(
                /*neg_a=*/false, a[kc], /*neg_b=*/false, b,
                /*c_mod=*/(short)0, acc, /*reuse_a=*/false, /*reuse_b=*/false);
        }
        // C/D layout: VGPR r, lanes 0-15 -> (M=r, N=lane); 16-31 -> (M=8+r).
        const int       n     = nt * 16 + (lane & 15);
        const long long rbase = m0 + ((lane >> 4) << 3);
        #pragma unroll
        for (int r = 0; r < 8; ++r)
            out[(rbase + r) * DIM + n] = acc[r];
    }
}

// ---------------------------------------------------------------------------
extern "C" void kernel_launch(void* const* d_in, const int* in_sizes, int n_in,
                              void* d_out, int out_size, void* d_ws, size_t ws_size,
                              hipStream_t stream) {
    const float* x    = (const float*)d_in[0];
    const int*   erow = (const int*)  d_in[1];
    const int*   ecol = (const int*)  d_in[2];
    const float* eval = (const float*)d_in[3];
    const float* W    = (const float*)d_in[4];
    float*       out  = (float*)d_out;
    float*       pre  = (float*)d_ws;          // N*D floats = 64 MB scratch

    // 1) zero pre_embed scratch (N*D/4 float4 stores)
    gc_zero<<<(N_NODES * DIM / 4) / 256, 256, 0, stream>>>((float4*)pre);

    // 2) SpMM scatter: one wave per edge, 8 waves per block
    gc_spmm_scatter<<<N_EDGES / 8, 256, 0, stream>>>(x, erow, ecol, eval, pre);

    // 3) WMMA GEMM: 128 rows per block
    gc_gemm_wmma<<<N_NODES / 128, 256, 0, stream>>>(pre, W, out);
}
